// KACN_27874337751729
// MI455X (gfx1250) — matrix operand order repaired and checked by reading, not compiled, not run on hardware
//
#include <hip/hip_runtime.h>
#include <hip/hip_bf16.h>

typedef _Float16 half8 __attribute__((ext_vector_type(8)));
typedef _Float16 v16h  __attribute__((ext_vector_type(16)));
typedef float    v8f   __attribute__((ext_vector_type(8)));
typedef float    f4    __attribute__((ext_vector_type(4)));

#define BROWS 16384
#define IN0   784
#define HID   1024
#define OUT2  10
#define K0DIM (IN0*4)   // 3136
#define K1DIM (HID*4)   // 4096

#define SCALE0 512.0f
#define SCALE1 512.0f

#define LDSPAD 40   // halves per LDS row (80B: 16B-aligned, 20-bank stride -> conflict-free)

union AFrag { v16h v; half8 h[2]; };
union H16   { half8 v[2]; _Float16 s[16]; };

// gfx1250 has a hardware V_TANH_F32 TRANS op — avoid the branchy libm tanhf.
__device__ __forceinline__ float fast_tanh(float x) {
#if __has_builtin(__builtin_amdgcn_tanhf)
    return __builtin_amdgcn_tanhf(x);
#elif __has_builtin(__builtin_amdgcn_tanh_f32)
    return __builtin_amdgcn_tanh_f32(x);
#else
    float r;
    asm volatile("v_tanh_f32 %0, %1\n\tv_nop" : "=v"(r) : "v"(x));
    return r;
#endif
}

// T_0..T_3 of tanh(x): {1, t, 2t^2-1, 4t^3-3t}
__device__ __forceinline__ void cheby4(float x, _Float16* o) {
    float t  = fast_tanh(x);
    float t2 = t * t;
    o[0] = (_Float16)1.0f;
    o[1] = (_Float16)t;
    o[2] = (_Float16)(2.0f * t2 - 1.0f);
    o[3] = (_Float16)(t * (4.0f * t2 - 3.0f));
}

// coeffs0 (784,1024,4) f32  ->  W0t[n][k] f16 * SCALE0, k = i*4+d, row stride K0DIM
__global__ __launch_bounds__(256) void prep_w0(const float* __restrict__ c0,
                                               _Float16* __restrict__ w0t) {
    int idx = blockIdx.x * 256 + threadIdx.x;           // exact multiple, no guard
    int n = idx / K0DIM, k = idx % K0DIM;
    int i = k >> 2, d = k & 3;
    w0t[idx] = (_Float16)(c0[((size_t)i * HID + n) * 4 + d] * SCALE0);
}

// coeffs1 (1024,10,4) f32 -> W1t[n][k] f16 * SCALE1, n padded to 16 with zeros
__global__ __launch_bounds__(256) void prep_w1(const float* __restrict__ c1,
                                               _Float16* __restrict__ w1t) {
    int idx = blockIdx.x * 256 + threadIdx.x;
    int n = idx / K1DIM, k = idx % K1DIM;
    int i = k >> 2, d = k & 3;
    float v = (n < OUT2) ? c1[((size_t)i * OUT2 + n) * 4 + d] * SCALE1 : 0.0f;
    w1t[idx] = (_Float16)v;
}

// GEMM1: h[16384,1024] = Basis(x)[16384,3136] * W0t^T, fused basis computation.
// Macro tile 128(M) x 128(N), 8 waves in 4x2 grid, wave tile 32x64 (2x4 WMMA).
// Double-buffered LDS: one barrier/iter, global loads overlap WMMA.
__global__ __launch_bounds__(256) void gemm1(const float* __restrict__ x,
                                             const _Float16* __restrict__ w0t,
                                             float* __restrict__ hbuf) {
    __shared__ _Float16 As[2][128 * LDSPAD];   // [m][k] 128x32 (padded)
    __shared__ _Float16 Bs[2][128 * LDSPAD];   // [n][k] 128x32 (padded)

    const int tid  = threadIdx.x;
    const int m0   = blockIdx.x * 128;
    const int n0   = blockIdx.y * 128;
    const int lane = tid & 31;
    const int wid  = tid >> 5;
    const int wm   = wid & 3;          // 4 wave rows
    const int wn   = wid >> 2;         // 2 wave cols
    const int lg   = lane >> 4;        // half-wave group
    const int lm   = lane & 15;

    v8f acc[2][4];
#pragma unroll
    for (int a = 0; a < 2; ++a)
#pragma unroll
        for (int b = 0; b < 4; ++b) acc[a][b] = (v8f)(0.0f);

    const int am = tid >> 1;                 // A loader: row 0..127
    const int ac = (tid & 1) * 4;            // channel offset 0 or 4 (of 8)
    const int bn = tid >> 1;                 // B loader: n row 0..127
    const int bc = tid & 1;                  // which 16-half chunk of 32

    const float* xrow = x + (size_t)(m0 + am) * IN0 + ac;
    const _Float16* brow = w0t + (size_t)(n0 + bn) * K0DIM + bc * 16;

    // ---- prologue: tile 0 -> buffer 0 ----
    {
        f4 xv = *(const f4*)(xrow);
        H16 w;
#pragma unroll
        for (int j = 0; j < 4; ++j) cheby4(xv[j], &w.s[j * 4]);
        *(half8*)&As[0][am * LDSPAD + (tid & 1) * 16]     = w.v[0];
        *(half8*)&As[0][am * LDSPAD + (tid & 1) * 16 + 8] = w.v[1];
        *(half8*)&Bs[0][bn * LDSPAD + bc * 16]     = *(const half8*)(brow);
        *(half8*)&Bs[0][bn * LDSPAD + bc * 16 + 8] = *(const half8*)(brow + 8);
    }
    __syncthreads();

    const int NT = K0DIM / 32;   // 98
    for (int it = 0; it < NT; ++it) {
        const int cur = it & 1;
        const int nxt = cur ^ 1;
        const bool more = (it + 1 < NT);

        // ---- issue next tile's global loads + basis compute early ----
        H16 w;
        half8 nb0, nb1;
        if (more) {
            const int kk = (it + 1) * 32;
            f4 xv = *(const f4*)(xrow + (kk >> 2));
#pragma unroll
            for (int j = 0; j < 4; ++j) cheby4(xv[j], &w.s[j * 4]);
            nb0 = *(const half8*)(brow + kk);
            nb1 = *(const half8*)(brow + kk + 8);
        }

        // ---- fragments in exact ISA VGPR layout ----
        AFrag a[2], b[4];
#pragma unroll
        for (int tm = 0; tm < 2; ++tm) {
            int row = wm * 32 + tm * 16 + lm;
            a[tm].h[0] = *(const half8*)&As[cur][row * LDSPAD + lg * 8];        // k 0-7 / 8-15
            a[tm].h[1] = *(const half8*)&As[cur][row * LDSPAD + 16 + lg * 8];   // k 16-23 / 24-31
        }
#pragma unroll
        for (int tn = 0; tn < 4; ++tn) {
            int col = wn * 64 + tn * 16 + lm;
            b[tn].h[0] = *(const half8*)&Bs[cur][col * LDSPAD + lg * 8];        // k 0-7 / 8-15
            b[tn].h[1] = *(const half8*)&Bs[cur][col * LDSPAD + 16 + lg * 8];   // k 16-23 / 24-31
        }
#pragma unroll
        for (int tm = 0; tm < 2; ++tm)
#pragma unroll
            for (int tn = 0; tn < 4; ++tn)
                acc[tm][tn] = __builtin_amdgcn_wmma_f32_16x16x32_f16(
                    false, a[tm].v, false, b[tn].v, (short)0, acc[tm][tn], false, false);

        if (more) {
            *(half8*)&As[nxt][am * LDSPAD + (tid & 1) * 16]     = w.v[0];
            *(half8*)&As[nxt][am * LDSPAD + (tid & 1) * 16 + 8] = w.v[1];
            *(half8*)&Bs[nxt][bn * LDSPAD + bc * 16]     = nb0;
            *(half8*)&Bs[nxt][bn * LDSPAD + bc * 16 + 8] = nb1;
            __syncthreads();
        }
    }

    const float inv = 1.0f / SCALE0;
#pragma unroll
    for (int tm = 0; tm < 2; ++tm)
#pragma unroll
        for (int tn = 0; tn < 4; ++tn)
#pragma unroll
            for (int r = 0; r < 8; ++r) {
                int m = m0 + wm * 32 + tm * 16 + r + lg * 8;
                int n = n0 + wn * 64 + tn * 16 + lm;
                hbuf[(size_t)m * HID + n] = acc[tm][tn][r] * inv;
            }
}

// GEMM2: out[16384,10] = Basis(h)[16384,4096] * W1t^T (N padded to 16), fused basis.
// Macro tile 128(M) x 16(N), 8 waves each one 16x16 WMMA tile. Double-buffered.
__global__ __launch_bounds__(256) void gemm2(const float* __restrict__ hbuf,
                                             const _Float16* __restrict__ w1t,
                                             float* __restrict__ out) {
    __shared__ _Float16 As[2][128 * LDSPAD];
    __shared__ _Float16 Bs[2][16 * LDSPAD];

    const int tid  = threadIdx.x;
    const int m0   = blockIdx.x * 128;
    const int lane = tid & 31;
    const int wid  = tid >> 5;
    const int lg   = lane >> 4;
    const int lm   = lane & 15;

    v8f acc = (v8f)(0.0f);

    const int am = tid >> 1;
    const int ac = (tid & 1) * 4;
    const float* hrow = hbuf + (size_t)(m0 + am) * HID + ac;
    const bool bldr = (tid < 64);
    const int  bn   = tid >> 2;        // n row 0..15
    const int  bcw  = tid & 3;         // 8-half chunk
    const _Float16* brow = w1t + (size_t)bn * K1DIM + bcw * 8;

    // ---- prologue ----
    {
        f4 hv = *(const f4*)(hrow);
        H16 w;
#pragma unroll
        for (int j = 0; j < 4; ++j) cheby4(hv[j], &w.s[j * 4]);
        *(half8*)&As[0][am * LDSPAD + (tid & 1) * 16]     = w.v[0];
        *(half8*)&As[0][am * LDSPAD + (tid & 1) * 16 + 8] = w.v[1];
        if (bldr) *(half8*)&Bs[0][bn * LDSPAD + bcw * 8] = *(const half8*)(brow);
    }
    __syncthreads();

    const int NT = K1DIM / 32;   // 128
    for (int it = 0; it < NT; ++it) {
        const int cur = it & 1;
        const int nxt = cur ^ 1;
        const bool more = (it + 1 < NT);

        H16 w;
        half8 nb0;
        if (more) {
            const int kk = (it + 1) * 32;
            f4 hv = *(const f4*)(hrow + (kk >> 2));
#pragma unroll
            for (int j = 0; j < 4; ++j) cheby4(hv[j], &w.s[j * 4]);
            if (bldr) nb0 = *(const half8*)(brow + kk);
        }

        AFrag a, b;
        int row = wid * 16 + lm;
        a.h[0] = *(const half8*)&As[cur][row * LDSPAD + lg * 8];
        a.h[1] = *(const half8*)&As[cur][row * LDSPAD + 16 + lg * 8];
        b.h[0] = *(const half8*)&Bs[cur][lm * LDSPAD + lg * 8];
        b.h[1] = *(const half8*)&Bs[cur][lm * LDSPAD + 16 + lg * 8];

        acc = __builtin_amdgcn_wmma_f32_16x16x32_f16(
            false, a.v, false, b.v, (short)0, acc, false, false);

        if (more) {
            *(half8*)&As[nxt][am * LDSPAD + (tid & 1) * 16]     = w.v[0];
            *(half8*)&As[nxt][am * LDSPAD + (tid & 1) * 16 + 8] = w.v[1];
            if (bldr) *(half8*)&Bs[nxt][bn * LDSPAD + bcw * 8] = nb0;
            __syncthreads();
        }
    }

    const float inv = 1.0f / SCALE1;
    if (lm < OUT2) {
#pragma unroll
        for (int r = 0; r < 8; ++r) {
            int m = m0 + wid * 16 + r + lg * 8;
            out[(size_t)m * OUT2 + lm] = acc[r] * inv;
        }
    }
}

extern "C" void kernel_launch(void* const* d_in, const int* in_sizes, int n_in,
                              void* d_out, int out_size, void* d_ws, size_t ws_size,
                              hipStream_t stream) {
    const float* x  = (const float*)d_in[0];   // (16384, 784)
    const float* c0 = (const float*)d_in[1];   // (784, 1024, 4)
    const float* c1 = (const float*)d_in[2];   // (1024, 10, 4)
    float* out = (float*)d_out;                // (16384, 10)

    char* ws = (char*)d_ws;
    const size_t w0_bytes = (size_t)HID * K0DIM * sizeof(_Float16);   // 6,422,528
    const size_t w1_bytes = (size_t)16 * K1DIM * sizeof(_Float16);    //   131,072
    _Float16* w0t  = (_Float16*)ws;
    _Float16* w1t  = (_Float16*)(ws + w0_bytes);
    float*    hbuf = (float*)(ws + w0_bytes + w1_bytes);              // 67,108,864

    prep_w0<<<(HID * K0DIM) / 256, 256, 0, stream>>>(c0, w0t);
    prep_w1<<<(16 * K1DIM) / 256, 256, 0, stream>>>(c1, w1t);

    dim3 g1(BROWS / 128, HID / 128);   // 128 x 8 = 1024 workgroups
    gemm1<<<g1, 256, 0, stream>>>(x, w0t, hbuf);

    gemm2<<<BROWS / 128, 256, 0, stream>>>(hbuf, w1t, out);
}
